// PAMNet_18459769438710
// MI455X (gfx1250) — matrix operand order repaired and checked by reading, not compiled, not run on hardware
//
#include <hip/hip_runtime.h>
#include <hip/hip_bf16.h>
#include <math.h>

// ---------------- problem constants (from reference) ----------------
#define N_NODES 20000
#define E_G     200000
#define E_L     100000
#define N_T     300000
#define N_P     300000
#define DIMF    156
#define NR_     16
#define NS_     7
#define NRAD_   6
#define PI_F    3.14159265358979323846f

typedef __attribute__((ext_vector_type(16))) _Float16 v16h;
typedef __attribute__((ext_vector_type(8)))  float    v8f;

// ---------------- weight prep: W[K x N] f32 -> W^T padded f16 [Npad x Kp] ----
__global__ void pamnet_prep_wT(const float* __restrict__ W, _Float16* __restrict__ Wt,
                               int K, int N, int Kp, int Npad) {
    int g = blockIdx.x * blockDim.x + threadIdx.x;
    if (g >= Npad * Kp) return;
    int n = g / Kp, k = g - n * Kp;
    float v = (k < K && n < N) ? W[(size_t)k * N + n] : 0.f;
    Wt[g] = (_Float16)v;
}

// ---------------- edge features: 16 bessel + 3 attr -> f16 [E x 32] ---------
__device__ __forceinline__ float pamnet_env(float xq) {
    float xs = fmaxf(xq, 1e-6f);
    float x2 = xs * xs, x4 = x2 * x2, x5 = x4 * xs, x6 = x5 * xs, x7 = x6 * xs;
    float e = 1.f / xs - 28.f * x5 + 48.f * x6 - 21.f * x7;
    return (xq < 1.f) ? e : 0.f;
}

__global__ void pamnet_edge_feat(const float* __restrict__ pos, const int* __restrict__ eidx,
                                 int E, const float* __restrict__ attr, float cutoff,
                                 _Float16* __restrict__ feat, float* __restrict__ distOut) {
    int e = blockIdx.x * blockDim.x + threadIdx.x;
    if (e >= E) return;
    int j = eidx[e], i = eidx[E + e];
    float dx = pos[i * 3 + 0] - pos[j * 3 + 0];
    float dy = pos[i * 3 + 1] - pos[j * 3 + 1];
    float dz = pos[i * 3 + 2] - pos[j * 3 + 2];
    float d = sqrtf(dx * dx + dy * dy + dz * dz + 1e-12f);
    if (distOut) distOut[e] = d;
    float xq = d / cutoff;
    float env = pamnet_env(xq);
    _Float16* f = feat + (size_t)e * 32;
#pragma unroll
    for (int n = 1; n <= NR_; ++n)
        f[n - 1] = (_Float16)(env * sinf((float)n * PI_F * xq));
    f[16] = (_Float16)attr[(size_t)e * 3 + 0];
    f[17] = (_Float16)attr[(size_t)e * 3 + 1];
    f[18] = (_Float16)attr[(size_t)e * 3 + 2];
#pragma unroll
    for (int k = 19; k < 32; ++k) f[k] = (_Float16)0.f;
}

// ---------------- spherical basis features: 42 (=7x6) -> f16 [T x 48] -------
// swapMode 0 (triplet): center=jl[b], q1=il[b];  swapMode 1 (pair): center=il[b], q1=jl[b]
__global__ void pamnet_sbf_feat(const float* __restrict__ pos,
                                const int* __restrict__ jl, const int* __restrict__ il,
                                const float* __restrict__ dist_l,
                                const int* __restrict__ idxA, const int* __restrict__ idxB,
                                int swapMode, int T, _Float16* __restrict__ feat) {
    int t = blockIdx.x * blockDim.x + threadIdx.x;
    if (t >= T) return;
    int a = idxA[t], b = idxB[t];
    int c  = swapMode ? il[b] : jl[b];
    int q1 = swapMode ? jl[b] : il[b];
    int q2 = jl[a];
    float v1x = pos[q1 * 3 + 0] - pos[c * 3 + 0];
    float v1y = pos[q1 * 3 + 1] - pos[c * 3 + 1];
    float v1z = pos[q1 * 3 + 2] - pos[c * 3 + 2];
    float v2x = pos[q2 * 3 + 0] - pos[c * 3 + 0];
    float v2y = pos[q2 * 3 + 1] - pos[c * 3 + 1];
    float v2z = pos[q2 * 3 + 2] - pos[c * 3 + 2];
    float cx = v1y * v2z - v1z * v2y;
    float cy = v1z * v2x - v1x * v2z;
    float cz = v1x * v2y - v1y * v2x;
    float cn = sqrtf(cx * cx + cy * cy + cz * cz);
    float dt = v1x * v2x + v1y * v2y + v1z * v2z;
    float ang = atan2f(cn + 1e-12f, dt);
    float d = dist_l[a];
    float xq = d / 2.6f;
    float env = pamnet_env(xq);
    float rad[NRAD_];
#pragma unroll
    for (int r = 0; r < NRAD_; ++r)
        rad[r] = env * sinf((float)(r + 1) * PI_F * xq);
    _Float16* f = feat + (size_t)t * 48;
#pragma unroll
    for (int s = 0; s < NS_; ++s) {
        float cs = cosf((float)s * ang);
#pragma unroll
        for (int r = 0; r < NRAD_; ++r)
            f[s * NRAD_ + r] = (_Float16)(cs * rad[r]);
    }
#pragma unroll
    for (int k = NS_ * NRAD_; k < 48; ++k) f[k] = (_Float16)0.f;
}

// ---------------- elementwise / scatter helpers -----------------------------
__global__ void pamnet_zero_f32(float* __restrict__ p, int n) {
    int i = blockIdx.x * blockDim.x + threadIdx.x;
    if (i < n) p[i] = 0.f;
}

__global__ void pamnet_seed_mlacc(const _Float16* __restrict__ ml, const float* __restrict__ el,
                                  float* __restrict__ acc, int n) {
    int i = blockIdx.x * blockDim.x + threadIdx.x;
    if (i < n) acc[i] = (float)ml[i] * el[i];
}

__global__ void pamnet_trip_scatter(const _Float16* __restrict__ ml, const _Float16* __restrict__ sbf,
                                    const int* __restrict__ srcIdx, const int* __restrict__ dstIdx,
                                    int T, float* __restrict__ acc) {
    int g = blockIdx.x * blockDim.x + threadIdx.x;
    if (g >= T * DIMF) return;
    int t = g / DIMF, c = g - t * DIMF;
    float v = (float)ml[(size_t)srcIdx[t] * DIMF + c] * (float)sbf[(size_t)t * DIMF + c];
    atomicAdd(&acc[(size_t)dstIdx[t] * DIMF + c], v);
}

__global__ void pamnet_edge_scatter(const float* __restrict__ acc, const int* __restrict__ il,
                                    int E, float* __restrict__ nodeacc) {
    int g = blockIdx.x * blockDim.x + threadIdx.x;
    if (g >= E * DIMF) return;
    int e = g / DIMF, c = g - e * DIMF;
    atomicAdd(&nodeacc[(size_t)il[e] * DIMF + c], acc[g]);
}

// ---------------- gather-fused WMMA GEMM ------------------------------------
// A row = concat of up to 3 segments (each optionally index-gathered, f32 or f16),
// zero padded to Kp. Each block: one 16-row tile; wave w: output cols [16w,16w+16).
// Wt: f16 [Npad x Kp], row n = output column n of W (i.e. W^T), K padded to Kp.
// Epilogue: v = (acc + bias); optional SiLU; optional + resid; then either
// direct store (f32 and/or f16) or fp32 atomic scatter-add into outF[scatterIdx[row]].
__device__ __forceinline__ float pamnet_load_seg(const void* p, const int* idx, int dt,
                                                 int row, int k, int w) {
    int r = idx ? idx[row] : row;
    if (dt) return (float)((const _Float16*)p)[(size_t)r * w + k];
    return ((const float*)p)[(size_t)r * w + k];
}

__global__ __launch_bounds__(320)
void pamnet_gemm_wmma(const void* s0, const int* i0, int w0, int t0,
                      const void* s1, const int* i1, int w1, int t1,
                      const void* s2, const int* i2, int w2, int t2,
                      const _Float16* __restrict__ Wt, const float* __restrict__ bias,
                      int M, int Nout, int Kp, int doSilu,
                      float* outF, _Float16* outH,
                      const float* resid, const int* scatterIdx) {
    __shared__ _Float16 At[16 * 480];
    const int tile = blockIdx.x;
    const int tid  = threadIdx.x;
    const int w01 = w0 + w1, w012 = w01 + w2;
    const int total = 16 * Kp;
    for (int e = tid; e < total; e += blockDim.x) {
        int r = e / Kp, k = e - r * Kp;
        int row = tile * 16 + r;
        float v = 0.f;
        if (row < M && k < w012) {
            if (k < w0)       v = pamnet_load_seg(s0, i0, t0, row, k, w0);
            else if (k < w01) v = pamnet_load_seg(s1, i1, t1, row, k - w0, w1);
            else              v = pamnet_load_seg(s2, i2, t2, row, k - w01, w2);
        }
        At[e] = (_Float16)v;
    }
    __syncthreads();

    const int wave = tid >> 5, lane = tid & 31;
    const int nBase = wave << 4;
    const int lm = lane & 15;
    const int hi = lane >> 4;  // 0 or 1 (lane half)
    v8f acc = {};
    // A layout: lane half selects K base (+16),
    // 16 contiguous K per lane (matches B).
    const _Float16* Arow = &At[lm * Kp + (hi << 4)];
    // B: lane = column, lane half selects K +16, 16 contiguous K per lane.
    const _Float16* Brow = &Wt[(size_t)(nBase + lm) * Kp + (hi << 4)];
    for (int kk = 0; kk < Kp; kk += 32) {
        v16h a;
        const _Float16* ap = Arow + kk;
#pragma unroll
        for (int r = 0; r < 4; ++r) {
            a[2 * r]     = ap[2 * r];
            a[2 * r + 1] = ap[2 * r + 1];
            a[8 + 2 * r] = ap[8 + 2 * r];
            a[9 + 2 * r] = ap[9 + 2 * r];
        }
        v16h b = *(const v16h*)(Brow + kk);  // 32B aligned: Kp multiple of 16
        acc = __builtin_amdgcn_wmma_f32_16x16x32_f16(false, a, false, b,
                                                     (short)0, acc, false, false);
    }
    const int n = nBase + lm;
    const int rowBase = tile * 16 + (hi << 3);
    if (n < Nout) {
#pragma unroll
        for (int r = 0; r < 8; ++r) {
            int row = rowBase + r;
            if (row < M) {
                float v = acc[r] + bias[n];
                if (doSilu) v = v / (1.f + expf(-v));
                if (resid)  v += resid[(size_t)row * Nout + n];
                if (scatterIdx) {
                    atomicAdd(outF + (size_t)scatterIdx[row] * Nout + n, v);
                } else {
                    if (outF) outF[(size_t)row * Nout + n] = v;
                    if (outH) outH[(size_t)row * Nout + n] = (_Float16)v;
                }
            }
        }
    }
}

// ---------------- host orchestration ----------------------------------------
extern "C" void kernel_launch(void* const* d_in, const int* in_sizes, int n_in,
                              void* d_out, int out_size, void* d_ws, size_t ws_size,
                              hipStream_t stream) {
    (void)in_sizes; (void)n_in; (void)out_size; (void)ws_size;

    const float* pos      = (const float*)d_in[0];
    const float* x        = (const float*)d_in[1];
    const float* seq      = (const float*)d_in[2];
    const float* eattr_g  = (const float*)d_in[3];
    const float* eattr_l  = (const float*)d_in[4];
    const int*   eig      = (const int*)d_in[5];
    const int*   eil      = (const int*)d_in[6];
    const int*   idx_kj   = (const int*)d_in[7];
    const int*   idx_ji   = (const int*)d_in[8];
    const int*   idx_jj_p = (const int*)d_in[9];
    const int*   idx_ji_p = (const int*)d_in[10];
    const float* W_rbf_g  = (const float*)d_in[11];
    const float* b_rbf_g  = (const float*)d_in[12];
    const float* W_rbf_l  = (const float*)d_in[13];
    const float* b_rbf_l  = (const float*)d_in[14];
    const float* W_sbf1   = (const float*)d_in[15];
    const float* b_sbf1   = (const float*)d_in[16];
    const float* W_sbf2   = (const float*)d_in[17];
    const float* b_sbf2   = (const float*)d_in[18];
    const float* Wg1      = (const float*)d_in[19];
    const float* bg1      = (const float*)d_in[20];
    const float* Wgm      = (const float*)d_in[21];
    const float* bgm      = (const float*)d_in[22];
    const float* Wg2      = (const float*)d_in[23];
    const float* bg2      = (const float*)d_in[24];
    const float* Wlm      = (const float*)d_in[25];
    const float* blm      = (const float*)d_in[26];
    const float* Wl2      = (const float*)d_in[27];
    const float* bl2      = (const float*)d_in[28];
    const float* Ws       = (const float*)d_in[29];
    const float* bs       = (const float*)d_in[30];
    const float* Wo       = (const float*)d_in[31];
    const float* bo       = (const float*)d_in[32];

    const int* jg = eig;           const int* ig = eig + E_G;
    const int* jl = eil;           const int* il = eil + E_L;

    // ---- workspace layout (256B aligned) ----
    char* base = (char*)d_ws;
    size_t off = 0;
    auto alloc = [&](size_t bytes) -> char* {
        char* p = base + off;
        off = (off + bytes + 255) & ~(size_t)255;
        return p;
    };
    _Float16* wt_rbf_g = (_Float16*)alloc((size_t)160 * 32 * 2);
    _Float16* wt_rbf_l = (_Float16*)alloc((size_t)160 * 32 * 2);
    _Float16* wt_sbf1  = (_Float16*)alloc((size_t)160 * 48 * 2);
    _Float16* wt_sbf2  = (_Float16*)alloc((size_t)160 * 48 * 2);
    _Float16 *wt_g1[2], *wt_gm[2], *wt_g2[2], *wt_lm[2], *wt_l2[2];
    for (int t = 0; t < 2; ++t) {
        wt_g1[t] = (_Float16*)alloc((size_t)160 * 160 * 2);
        wt_gm[t] = (_Float16*)alloc((size_t)160 * 480 * 2);
        wt_g2[t] = (_Float16*)alloc((size_t)160 * 160 * 2);
        wt_lm[t] = (_Float16*)alloc((size_t)160 * 320 * 2);
        wt_l2[t] = (_Float16*)alloc((size_t)160 * 160 * 2);
    }
    _Float16* wt_s  = (_Float16*)alloc((size_t)64 * 320 * 2);
    _Float16* wt_o  = (_Float16*)alloc((size_t)16 * 128 * 2);
    _Float16* feat_g = (_Float16*)alloc((size_t)E_G * 32 * 2);
    _Float16* feat_l = (_Float16*)alloc((size_t)E_L * 32 * 2);
    float*    dist_l = (float*)   alloc((size_t)E_L * 4);
    _Float16* eg_h   = (_Float16*)alloc((size_t)E_G * DIMF * 2);
    float*    el_f   = (float*)   alloc((size_t)E_L * DIMF * 4);
    _Float16* sbff   = (_Float16*)alloc((size_t)N_T * 48 * 2);
    _Float16* sbf1_h = (_Float16*)alloc((size_t)N_T * DIMF * 2);
    _Float16* sbf2_h = (_Float16*)alloc((size_t)N_P * DIMF * 2);
    _Float16* h_h    = (_Float16*)alloc((size_t)N_NODES * DIMF * 2);
    float*    gacc   = (float*)   alloc((size_t)N_NODES * DIMF * 4);
    float*    xg_b   = (float*)   alloc((size_t)N_NODES * DIMF * 4);
    float*    xl_b   = (float*)   alloc((size_t)N_NODES * DIMF * 4);
    _Float16* ml_h   = (_Float16*)alloc((size_t)E_L * DIMF * 2);
    float*    mlacc  = (float*)   alloc((size_t)E_L * DIMF * 4);
    float*    nodeacc= (float*)   alloc((size_t)N_NODES * DIMF * 4);
    float*    structb= (float*)   alloc((size_t)N_NODES * 64 * 4);

    auto prep = [&](const float* W, _Float16* Wt, int K, int N, int Kp, int Npad) {
        int tot = Npad * Kp;
        pamnet_prep_wT<<<dim3((tot + 255) / 256), dim3(256), 0, stream>>>(W, Wt, K, N, Kp, Npad);
    };
    auto gemm = [&](const void* s0, const int* i0, int w0, int t0,
                    const void* s1, const int* i1, int w1, int t1,
                    const void* s2, const int* i2, int w2, int t2,
                    const _Float16* Wt, const float* bias,
                    int M, int Nout, int Kp, int doSilu,
                    float* outF, _Float16* outH, const float* resid, const int* sidx) {
        int nt = (Nout + 15) / 16;
        pamnet_gemm_wmma<<<dim3((M + 15) / 16), dim3(32 * nt), 0, stream>>>(
            s0, i0, w0, t0, s1, i1, w1, t1, s2, i2, w2, t2,
            Wt, bias, M, Nout, Kp, doSilu, outF, outH, resid, sidx);
    };
    auto zero = [&](float* p, int n) {
        pamnet_zero_f32<<<dim3((n + 255) / 256), dim3(256), 0, stream>>>(p, n);
    };

    // ---- weight prep ----
    prep(W_rbf_g, wt_rbf_g, NR_ + 3, DIMF, 32, 160);
    prep(W_rbf_l, wt_rbf_l, NR_ + 3, DIMF, 32, 160);
    prep(W_sbf1, wt_sbf1, NS_ * NRAD_, DIMF, 48, 160);
    prep(W_sbf2, wt_sbf2, NS_ * NRAD_, DIMF, 48, 160);
    for (int t = 0; t < 2; ++t) {
        prep(Wg1 + (size_t)t * DIMF * DIMF,     wt_g1[t], DIMF,     DIMF, 160, 160);
        prep(Wgm + (size_t)t * 3 * DIMF * DIMF, wt_gm[t], 3 * DIMF, DIMF, 480, 160);
        prep(Wg2 + (size_t)t * DIMF * DIMF,     wt_g2[t], DIMF,     DIMF, 160, 160);
        prep(Wlm + (size_t)t * 2 * DIMF * DIMF, wt_lm[t], 2 * DIMF, DIMF, 320, 160);
        prep(Wl2 + (size_t)t * DIMF * DIMF,     wt_l2[t], DIMF,     DIMF, 160, 160);
    }
    prep(Ws, wt_s, 2 * DIMF, 64, 320, 64);
    prep(Wo, wt_o, 128, 15, 128, 16);

    // ---- edge features + edge embeddings ----
    pamnet_edge_feat<<<dim3((E_G + 255) / 256), dim3(256), 0, stream>>>(
        pos, eig, E_G, eattr_g, 20.0f, feat_g, nullptr);
    pamnet_edge_feat<<<dim3((E_L + 255) / 256), dim3(256), 0, stream>>>(
        pos, eil, E_L, eattr_l, 2.6f, feat_l, dist_l);
    gemm(feat_g, nullptr, 32, 1, nullptr, nullptr, 0, 0, nullptr, nullptr, 0, 0,
         wt_rbf_g, b_rbf_g, E_G, DIMF, 32, 1, nullptr, eg_h, nullptr, nullptr);
    gemm(feat_l, nullptr, 32, 1, nullptr, nullptr, 0, 0, nullptr, nullptr, 0, 0,
         wt_rbf_l, b_rbf_l, E_L, DIMF, 32, 1, el_f, nullptr, nullptr, nullptr);

    // ---- spherical basis embeddings ----
    pamnet_sbf_feat<<<dim3((N_T + 255) / 256), dim3(256), 0, stream>>>(
        pos, jl, il, dist_l, idx_kj, idx_ji, 0, N_T, sbff);
    gemm(sbff, nullptr, 48, 1, nullptr, nullptr, 0, 0, nullptr, nullptr, 0, 0,
         wt_sbf1, b_sbf1, N_T, DIMF, 48, 1, nullptr, sbf1_h, nullptr, nullptr);
    pamnet_sbf_feat<<<dim3((N_P + 255) / 256), dim3(256), 0, stream>>>(
        pos, jl, il, dist_l, idx_jj_p, idx_ji_p, 1, N_P, sbff);
    gemm(sbff, nullptr, 48, 1, nullptr, nullptr, 0, 0, nullptr, nullptr, 0, 0,
         wt_sbf2, b_sbf2, N_P, DIMF, 48, 1, nullptr, sbf2_h, nullptr, nullptr);

    // ---- message passing layers ----
    const float* xg = x;
    const float* xl = x;
    for (int t = 0; t < 2; ++t) {
        // global branch
        gemm(xg, nullptr, DIMF, 0, nullptr, nullptr, 0, 0, nullptr, nullptr, 0, 0,
             wt_g1[t], bg1 + (size_t)t * DIMF, N_NODES, DIMF, 160, 1,
             nullptr, h_h, nullptr, nullptr);
        zero(gacc, N_NODES * DIMF);
        // m = silu(concat[h[jg], h[ig], eg] @ Wgm + bgm); scatter-add into gacc[ig]
        gemm(h_h, jg, DIMF, 1, h_h, ig, DIMF, 1, eg_h, nullptr, DIMF, 1,
             wt_gm[t], bgm + (size_t)t * DIMF, E_G, DIMF, 480, 1,
             gacc, nullptr, nullptr, ig);
        gemm(gacc, nullptr, DIMF, 0, nullptr, nullptr, 0, 0, nullptr, nullptr, 0, 0,
             wt_g2[t], bg2 + (size_t)t * DIMF, N_NODES, DIMF, 160, 1,
             xg_b, nullptr, xg, nullptr);
        // local branch
        gemm(xl, jl, DIMF, 0, xl, il, DIMF, 0, nullptr, nullptr, 0, 0,
             wt_lm[t], blm + (size_t)t * DIMF, E_L, DIMF, 320, 1,
             nullptr, ml_h, nullptr, nullptr);
        pamnet_seed_mlacc<<<dim3((E_L * DIMF + 255) / 256), dim3(256), 0, stream>>>(
            ml_h, el_f, mlacc, E_L * DIMF);
        pamnet_trip_scatter<<<dim3((N_T * DIMF + 255) / 256), dim3(256), 0, stream>>>(
            ml_h, sbf1_h, idx_kj, idx_ji, N_T, mlacc);
        pamnet_trip_scatter<<<dim3((N_P * DIMF + 255) / 256), dim3(256), 0, stream>>>(
            ml_h, sbf2_h, idx_jj_p, idx_ji_p, N_P, mlacc);
        zero(nodeacc, N_NODES * DIMF);
        pamnet_edge_scatter<<<dim3((E_L * DIMF + 255) / 256), dim3(256), 0, stream>>>(
            mlacc, il, E_L, nodeacc);
        gemm(nodeacc, nullptr, DIMF, 0, nullptr, nullptr, 0, 0, nullptr, nullptr, 0, 0,
             wt_l2[t], bl2 + (size_t)t * DIMF, N_NODES, DIMF, 160, 1,
             xl_b, nullptr, xl, nullptr);
        xg = xg_b;
        xl = xl_b;
    }

    // ---- heads ----
    gemm(xg, nullptr, DIMF, 0, xl, nullptr, DIMF, 0, nullptr, nullptr, 0, 0,
         wt_s, bs, N_NODES, 64, 320, 0, structb, nullptr, nullptr, nullptr);
    gemm(seq, nullptr, 64, 0, structb, nullptr, 64, 0, nullptr, nullptr, 0, 0,
         wt_o, bo, N_NODES, 15, 128, 0, (float*)d_out, nullptr, nullptr, nullptr);
}
